// ResidualBlock_65566970740820
// MI455X (gfx1250) — matrix operand order repaired and checked by reading, not compiled, not run on hardware
//
#include <hip/hip_runtime.h>
#include <hip/hip_bf16.h>

// ---------------- problem constants ----------------
#define D_MODEL 2048
#define D_INNER 4096
#define D_CONV  4
#define D_STATE 16
#define DT_RANK 128
#define B_SZ    2
#define SEQ     2048
#define NROWS   (B_SZ * SEQ)            // 4096 flattened (b,l) rows
#define XDBL_N  (DT_RANK + 2 * D_STATE) // 160

typedef __attribute__((ext_vector_type(16))) __bf16 v16bf;
typedef __attribute__((ext_vector_type(8)))  __bf16 v8bf;
typedef __attribute__((ext_vector_type(8)))  float  v8f;

__device__ __forceinline__ __bf16 f2bf(float f) {
    union { float f; unsigned u; } a; a.f = f;
    unsigned r = a.u + 0x7FFFu + ((a.u >> 16) & 1u);   // round-to-nearest-even
    unsigned short h = (unsigned short)(r >> 16);
    return __builtin_bit_cast(__bf16, h);
}

__device__ __forceinline__ float sigmoidf_(float x) { return 1.0f / (1.0f + __expf(-x)); }
__device__ __forceinline__ float softplusf_(float x) {
    return (x > 20.0f) ? x : log1pf(__expf(x));
}

// ---------------- fp32 -> bf16 convert ----------------
__global__ void cvt_bf16_kernel(const float* __restrict__ src, __bf16* __restrict__ dst, int n) {
    int i = blockIdx.x * blockDim.x + threadIdx.x;
    if (i < n) dst[i] = f2bf(src[i]);
}

// ---------------- RMSNorm -> bf16 ----------------
__global__ void rmsnorm_kernel(const float* __restrict__ x, const float* __restrict__ scale,
                               __bf16* __restrict__ xn) {
    const int row = blockIdx.x;
    const float* xr = x + (size_t)row * D_MODEL;
    float s = 0.f;
    for (int i = threadIdx.x; i < D_MODEL; i += blockDim.x) {
        float v = xr[i]; s += v * v;
    }
    s += __shfl_xor(s, 1, 32); s += __shfl_xor(s, 2, 32); s += __shfl_xor(s, 4, 32);
    s += __shfl_xor(s, 8, 32); s += __shfl_xor(s, 16, 32);
    __shared__ float sred[9];
    int wv = threadIdx.x >> 5, lane = threadIdx.x & 31;
    if (lane == 0) sred[wv] = s;
    __syncthreads();
    if (threadIdx.x == 0) {
        float t = 0.f;
        for (int i = 0; i < 8; ++i) t += sred[i];
        sred[8] = t;
    }
    __syncthreads();
    float inv = rsqrtf(sred[8] / (float)D_MODEL + 1e-5f);
    __bf16* xo = xn + (size_t)row * D_MODEL;
    for (int i = threadIdx.x; i < D_MODEL; i += blockDim.x)
        xo[i] = f2bf(xr[i] * inv * scale[i]);
}

// ---------------- generic bf16 WMMA GEMM: C[M,N] = A[M,K] * B[N,K]^T ----------------
// MODE 0: C = acc            (optionally also bf16 copy to Cbf)
// MODE 1: C = softplus(acc + ep[n])
// MODE 2: C = acc + ep[m*ldc + n]   (residual add)
// GUARD : compile-time; false => branch-free inner loop (all 4 N-subtiles valid)
// Inner loop is register double-buffered: WMMAs consume fragments loaded one
// K-step earlier, so VMEM loads overlap matrix ops (no loadcnt drain per WMMA).
template <int MODE, bool GUARD>
__global__ void wmma_gemm_kernel(const __bf16* __restrict__ A, const __bf16* __restrict__ B,
                                 float* __restrict__ C, int M, int N, int K,
                                 int lda, int ldb, int ldc,
                                 const float* __restrict__ ep,
                                 __bf16* __restrict__ Cbf, int ldcb) {
    const int wave = threadIdx.x >> 5;
    const int lane = threadIdx.x & 31;
    const int half = lane >> 4;        // 0/1
    const int lm   = lane & 15;        // 0..15
    const int tileM = blockIdx.y * 64 + wave * 16;
    const int tileN = blockIdx.x * 64;
    const int m = tileM + lm;          // A row this lane loads

    v8f acc0 = {}, acc1 = {}, acc2 = {}, acc3 = {};
    const bool n0 = !GUARD || (tileN +  0) < N;
    const bool n1 = !GUARD || (tileN + 16) < N;
    const bool n2 = !GUARD || (tileN + 32) < N;
    const bool n3 = !GUARD || (tileN + 48) < N;

    const __bf16* rowA  = A + (size_t)m * lda + half * 8;
    const __bf16* rowB0 = B + (size_t)(tileN +  0 + lm) * ldb + half * 8;
    const __bf16* rowB1 = B + (size_t)(tileN + 16 + lm) * ldb + half * 8;
    const __bf16* rowB2 = B + (size_t)(tileN + 32 + lm) * ldb + half * 8;
    const __bf16* rowB3 = B + (size_t)(tileN + 48 + lm) * ldb + half * 8;

    auto ldA = [&](int kb) -> v16bf {
        v8bf lo = *(const v8bf*)(rowA + kb);
        v8bf hi = *(const v8bf*)(rowA + kb + 16);
        return __builtin_shufflevector(lo, hi,
            0, 1, 2, 3, 4, 5, 6, 7, 8, 9, 10, 11, 12, 13, 14, 15);
    };
    auto ldB = [&](const __bf16* row, int kb) -> v16bf {
        v8bf lo = *(const v8bf*)(row + kb);
        v8bf hi = *(const v8bf*)(row + kb + 16);
        return __builtin_shufflevector(lo, hi,
            0, 1, 2, 3, 4, 5, 6, 7, 8, 9, 10, 11, 12, 13, 14, 15);
    };

    // prologue: fragments for kb = 0
    v16bf aC = ldA(0);
    v16bf b0C = {}, b1C = {}, b2C = {}, b3C = {};
    if (n0) b0C = ldB(rowB0, 0);
    if (n1) b1C = ldB(rowB1, 0);
    if (n2) b2C = ldB(rowB2, 0);
    if (n3) b3C = ldB(rowB3, 0);

    #pragma unroll 2
    for (int kb = 0; kb + 32 < K; kb += 32) {
        const int kn = kb + 32;
        __builtin_prefetch((const void*)(rowA + kn + 32), 0, 1); // K tile after next
        // issue next-step loads (distinct live values -> no drain before WMMAs)
        v16bf aN = ldA(kn);
        v16bf b0N = {}, b1N = {}, b2N = {}, b3N = {};
        if (n0) b0N = ldB(rowB0, kn);
        if (n1) b1N = ldB(rowB1, kn);
        if (n2) b2N = ldB(rowB2, kn);
        if (n3) b3N = ldB(rowB3, kn);

        // matrix ops on fragments loaded one iteration ago
        if (n0) acc0 = __builtin_amdgcn_wmma_f32_16x16x32_bf16(
                    false, aC, false, b0C, (short)0, acc0, false, false);
        if (n1) acc1 = __builtin_amdgcn_wmma_f32_16x16x32_bf16(
                    false, aC, false, b1C, (short)0, acc1, false, false);
        if (n2) acc2 = __builtin_amdgcn_wmma_f32_16x16x32_bf16(
                    false, aC, false, b2C, (short)0, acc2, false, false);
        if (n3) acc3 = __builtin_amdgcn_wmma_f32_16x16x32_bf16(
                    false, aC, false, b3C, (short)0, acc3, false, false);

        aC = aN; b0C = b0N; b1C = b1N; b2C = b2N; b3C = b3N; // ping-pong
    }
    // epilogue K-step
    if (n0) acc0 = __builtin_amdgcn_wmma_f32_16x16x32_bf16(
                false, aC, false, b0C, (short)0, acc0, false, false);
    if (n1) acc1 = __builtin_amdgcn_wmma_f32_16x16x32_bf16(
                false, aC, false, b1C, (short)0, acc1, false, false);
    if (n2) acc2 = __builtin_amdgcn_wmma_f32_16x16x32_bf16(
                false, aC, false, b2C, (short)0, acc2, false, false);
    if (n3) acc3 = __builtin_amdgcn_wmma_f32_16x16x32_bf16(
                false, aC, false, b3C, (short)0, acc3, false, false);

#define STORE_TILE(T, ACC, OK)                                                     \
    if (OK) {                                                                      \
        int n = tileN + (T)*16 + lm;                                               \
        for (int v = 0; v < 8; ++v) {                                              \
            int mm = tileM + v + 8 * half;                                         \
            float val = ACC[v];                                                    \
            if (MODE == 1)      val = softplusf_(val + ep[n]);                     \
            else if (MODE == 2) val = val + ep[(size_t)mm * ldc + n];              \
            C[(size_t)mm * ldc + n] = val;                                         \
            if (Cbf) Cbf[(size_t)mm * ldcb + n] = f2bf(val);                       \
        }                                                                          \
    }
    STORE_TILE(0, acc0, n0)
    STORE_TILE(1, acc1, n1)
    STORE_TILE(2, acc2, n2)
    STORE_TILE(3, acc3, n3)
#undef STORE_TILE
}

// ---------------- depthwise causal conv (k=4) + bias + SiLU ----------------
// x_in[b,d,l] = xz[b,l,d]  (first D_INNER columns of xz); outputs (b,l,d) layout.
__global__ void conv_silu_kernel(const float* __restrict__ xz, const float* __restrict__ w,
                                 const float* __restrict__ bias,
                                 float* __restrict__ xc, __bf16* __restrict__ xcb) {
    int idx = blockIdx.x * blockDim.x + threadIdx.x;
    if (idx >= B_SZ * SEQ * D_INNER) return;
    int d = idx % D_INNER;
    int l = (idx / D_INNER) % SEQ;
    int b = idx / (D_INNER * SEQ);
    float acc = bias[d];
    #pragma unroll
    for (int k = 0; k < D_CONV; ++k) {
        int ls = l - (D_CONV - 1) + k;
        if (ls >= 0)
            acc += xz[((size_t)(b * SEQ + ls)) * (2 * D_INNER) + d] * w[d * D_CONV + k];
    }
    float y = acc * sigmoidf_(acc);
    size_t o = (size_t)(b * SEQ + l) * D_INNER + d;
    xc[o] = y;
    xcb[o] = f2bf(y);
}

// ---------------- selective scan (fused +u*D, *silu(z), bf16 out) ----------------
// one state per lane; 16-lane group per (b,d) channel; sequential over L.
__global__ void scan_kernel(const float* __restrict__ delta, const float* __restrict__ xdbl,
                            const float* __restrict__ xc, const float* __restrict__ xz,
                            const float* __restrict__ A_log, const float* __restrict__ Dv,
                            __bf16* __restrict__ ybf) {
    int tid = blockIdx.x * blockDim.x + threadIdx.x;
    int s  = tid & 15;
    int ch = tid >> 4;                // 0 .. B_SZ*D_INNER-1
    int b  = ch >> 12;                // / D_INNER
    int d  = ch & (D_INNER - 1);
    float Acoef = -__expf(A_log[d * D_STATE + s]);
    float Dd = Dv[d];
    float h = 0.f;
    for (int l = 0; l < SEQ; ++l) {
        size_t row = (size_t)(b * SEQ + l);
        float dt = delta[row * D_INNER + d];
        float u  = xc[row * D_INNER + d];
        float Bv = xdbl[row * XDBL_N + DT_RANK + s];
        float Cv = xdbl[row * XDBL_N + DT_RANK + D_STATE + s];
        h = h * __expf(dt * Acoef) + (dt * u) * Bv;
        float p = h * Cv;
        p += __shfl_xor(p, 1, 32);
        p += __shfl_xor(p, 2, 32);
        p += __shfl_xor(p, 4, 32);
        p += __shfl_xor(p, 8, 32);
        if (s == 0) {
            float y = p + u * Dd;
            float z = xz[row * (2 * D_INNER) + D_INNER + d];
            y *= z * sigmoidf_(z);
            ybf[row * D_INNER + d] = f2bf(y);
        }
    }
}

// ---------------- launcher ----------------
extern "C" void kernel_launch(void* const* d_in, const int* in_sizes, int n_in,
                              void* d_out, int out_size, void* d_ws, size_t ws_size,
                              hipStream_t stream) {
    const float* hs        = (const float*)d_in[0];
    const float* nsc       = (const float*)d_in[1];
    const float* in_proj_w = (const float*)d_in[2];
    const float* conv_w    = (const float*)d_in[3];
    const float* conv_b    = (const float*)d_in[4];
    const float* x_proj_w  = (const float*)d_in[5];
    const float* dt_proj_w = (const float*)d_in[6];
    const float* dt_proj_b = (const float*)d_in[7];
    const float* out_proj_w= (const float*)d_in[8];
    const float* A_log     = (const float*)d_in[9];
    const float* Dvec      = (const float*)d_in[10];

    float* out0 = (float*)d_out;                                   // (B,L,D_MODEL)
    float* xz   = out0 + (size_t)NROWS * D_MODEL;                  // (B,L,2*D_INNER)

    // workspace layout (256B aligned)
    char* ws = (char*)d_ws;
    size_t off = 0;
    auto alloc = [&](size_t bytes) -> char* {
        char* p = ws + off;
        off += (bytes + 255) & ~(size_t)255;
        return p;
    };
    __bf16* xn_bf   = (__bf16*)alloc((size_t)NROWS * D_MODEL * 2);
    __bf16* w_in_bf = (__bf16*)alloc((size_t)2 * D_INNER * D_MODEL * 2);
    float*  xc_f    = (float*) alloc((size_t)NROWS * D_INNER * 4);
    __bf16* xc_bf   = (__bf16*)alloc((size_t)NROWS * D_INNER * 2);
    __bf16* w_xp_bf = (__bf16*)alloc((size_t)XDBL_N * D_INNER * 2);
    float*  xdbl_f  = (float*) alloc((size_t)NROWS * XDBL_N * 4);
    __bf16* xdbl_bf = (__bf16*)alloc((size_t)NROWS * XDBL_N * 2);
    __bf16* w_dt_bf = (__bf16*)alloc((size_t)D_INNER * DT_RANK * 2);
    float*  delta_f = (float*) alloc((size_t)NROWS * D_INNER * 4);
    __bf16* y_bf    = (__bf16*)alloc((size_t)NROWS * D_INNER * 2);
    __bf16* w_op_bf = (__bf16*)alloc((size_t)D_MODEL * D_INNER * 2);

    // 1) weight conversions
    {
        int n;
        n = 2 * D_INNER * D_MODEL;
        cvt_bf16_kernel<<<(n + 255) / 256, 256, 0, stream>>>(in_proj_w, w_in_bf, n);
        n = XDBL_N * D_INNER;
        cvt_bf16_kernel<<<(n + 255) / 256, 256, 0, stream>>>(x_proj_w, w_xp_bf, n);
        n = D_INNER * DT_RANK;
        cvt_bf16_kernel<<<(n + 255) / 256, 256, 0, stream>>>(dt_proj_w, w_dt_bf, n);
        n = D_MODEL * D_INNER;
        cvt_bf16_kernel<<<(n + 255) / 256, 256, 0, stream>>>(out_proj_w, w_op_bf, n);
    }

    // 2) RMSNorm
    rmsnorm_kernel<<<NROWS, 256, 0, stream>>>(hs, nsc, xn_bf);

    // 3) xz = xn @ in_proj_w^T  -> directly into output region 1  (branch-free)
    {
        dim3 grid((2 * D_INNER) / 64, NROWS / 64);
        wmma_gemm_kernel<0, false><<<grid, 128, 0, stream>>>(xn_bf, w_in_bf, xz,
            NROWS, 2 * D_INNER, D_MODEL, D_MODEL, D_MODEL, 2 * D_INNER,
            nullptr, nullptr, 0);
    }

    // 4) depthwise causal conv + SiLU
    {
        int n = B_SZ * SEQ * D_INNER;
        conv_silu_kernel<<<(n + 255) / 256, 256, 0, stream>>>(xz, conv_w, conv_b, xc_f, xc_bf);
    }

    // 5) x_dbl = xc @ x_proj_w^T   (N=160 -> guarded variant, only 3 blocks wide)
    {
        dim3 grid((XDBL_N + 63) / 64, NROWS / 64);
        wmma_gemm_kernel<0, true><<<grid, 128, 0, stream>>>(xc_bf, w_xp_bf, xdbl_f,
            NROWS, XDBL_N, D_INNER, D_INNER, D_INNER, XDBL_N,
            nullptr, xdbl_bf, XDBL_N);
    }

    // 6) delta = softplus(dt_low @ dt_proj_w^T + dt_proj_b)  (branch-free)
    {
        dim3 grid(D_INNER / 64, NROWS / 64);
        wmma_gemm_kernel<1, false><<<grid, 128, 0, stream>>>(xdbl_bf, w_dt_bf, delta_f,
            NROWS, D_INNER, DT_RANK, XDBL_N, DT_RANK, D_INNER,
            dt_proj_b, nullptr, 0);
    }

    // 7) selective scan (fused skip, gate, bf16 out)
    {
        int nthreads = B_SZ * D_INNER * D_STATE; // 131072
        scan_kernel<<<nthreads / 256, 256, 0, stream>>>(delta_f, xdbl_f, xc_f, xz,
                                                        A_log, Dvec, y_bf);
    }

    // 8) out = y @ out_proj_w^T + residual  -> output region 0  (branch-free)
    {
        dim3 grid(D_MODEL / 64, NROWS / 64);
        wmma_gemm_kernel<2, false><<<grid, 128, 0, stream>>>(y_bf, w_op_bf, out0,
            NROWS, D_MODEL, D_INNER, D_INNER, D_INNER, D_MODEL,
            hs, nullptr, 0);
    }
    (void)in_sizes; (void)n_in; (void)out_size; (void)ws_size;
}
